// Linear_58669253264028
// MI455X (gfx1250) — matrix operand order repaired and checked by reading, not compiled, not run on hardware
//
#include <hip/hip_runtime.h>
#include <stdint.h>

typedef __bf16 v16bf __attribute__((ext_vector_type(16)));
typedef __bf16 v8bf  __attribute__((ext_vector_type(8)));
typedef float  v8f   __attribute__((ext_vector_type(8)));
typedef float  v4f   __attribute__((ext_vector_type(4)));
typedef unsigned int v4u __attribute__((ext_vector_type(4)));

#if defined(__has_builtin)
# if __has_builtin(__builtin_amdgcn_global_load_async_to_lds_b128)
#  define USE_ASYNC_LDS 1
# endif
#endif
#ifndef USE_ASYNC_LDS
# define USE_ASYNC_LDS 0
#endif

#if USE_ASYNC_LDS
// builtin parameter type per hipcc diagnostic:
//   '__attribute__((__vector_size__(4 * sizeof(int)))) int __device__ *'
typedef int async_v4i __attribute__((vector_size(4 * sizeof(int))));
# define ASYNC_G(p) ((__attribute__((address_space(1))) async_v4i*)(uintptr_t)(p))
# define ASYNC_L(p) ((__attribute__((address_space(3))) async_v4i*)(p))
# if __has_builtin(__builtin_amdgcn_s_wait_asynccnt)
#  define WAIT_ASYNC() __builtin_amdgcn_s_wait_asynccnt(0)
# else
#  define WAIT_ASYNC() asm volatile("s_wait_asynccnt 0" ::: "memory")
# endif
#else
# define WAIT_ASYNC() ((void)0)
#endif

// ---------------- precompute kernels ----------------

__global__ void zero_f4_kernel(v4f* __restrict__ p, long n4) {
  long i = (long)blockIdx.x * blockDim.x + threadIdx.x;
  v4f z = {0.f, 0.f, 0.f, 0.f};
  if (i < n4) p[i] = z;
}

// scatter spectrum through the algebraically-collapsed inverse Haar into
// deltaT[l][k] = delta_w[k][l]
__global__ void scatter_kernel(const int* __restrict__ idx,
                               const float* __restrict__ spec,
                               float* __restrict__ dT, int NF, int D) {
  int t = blockIdx.x * blockDim.x + threadIdx.x;
  if (t >= NF) return;
  const int h = D >> 1;
  const int r = idx[t], c = idx[NF + t];
  const float s = spec[t] * 0.1f;            // (1/sqrt2)^2 * (1/5)
  const int i = r & (h - 1), j = c & (h - 1);
  const int R = (r >= h), C = (c >= h);
#pragma unroll
  for (int p = 0; p < 2; ++p)
#pragma unroll
    for (int q = 0; q < 2; ++q) {
      float v = s;
      if (R && q) v = -v;
      if (C && p) v = -v;
      // transposed target: deltaT[2j+q][2i+p]
      atomicAdd(&dT[(long)(2 * j + q) * D + (2 * i + p)], v);
    }
}

// W_effT[l][k] = bf16(base_weight[l*D+k] + deltaT[l*D+k])  (pure elementwise)
__global__ void build_weff_kernel(const float* __restrict__ bw,
                                  const float* __restrict__ dT,
                                  unsigned short* __restrict__ wbf, long n8) {
  long t = (long)blockIdx.x * blockDim.x + threadIdx.x;
  if (t >= n8) return;
  const v4f* b4 = (const v4f*)bw;
  const v4f* d4 = (const v4f*)dT;
  v4f b0 = b4[2 * t], b1 = b4[2 * t + 1];
  v4f d0 = d4[2 * t], d1 = d4[2 * t + 1];
  v8bf o;
  o[0] = (__bf16)(b0[0] + d0[0]); o[1] = (__bf16)(b0[1] + d0[1]);
  o[2] = (__bf16)(b0[2] + d0[2]); o[3] = (__bf16)(b0[3] + d0[3]);
  o[4] = (__bf16)(b1[0] + d1[0]); o[5] = (__bf16)(b1[1] + d1[1]);
  o[6] = (__bf16)(b1[2] + d1[2]); o[7] = (__bf16)(b1[3] + d1[3]);
  *((v8bf*)wbf + t) = o;
}

__global__ void cvt_x_kernel(const float* __restrict__ x,
                             unsigned short* __restrict__ xb, long n8) {
  long t = (long)blockIdx.x * blockDim.x + threadIdx.x;
  if (t >= n8) return;
  const v4f* x4 = (const v4f*)x;
  v4f a = x4[2 * t], b = x4[2 * t + 1];
  v8bf o;
  o[0] = (__bf16)a[0]; o[1] = (__bf16)a[1]; o[2] = (__bf16)a[2]; o[3] = (__bf16)a[3];
  o[4] = (__bf16)b[0]; o[5] = (__bf16)b[1]; o[6] = (__bf16)b[2]; o[7] = (__bf16)b[3];
  *((v8bf*)xb + t) = o;
}

// ---------------- WMMA GEMM ----------------
// C[M,N] = A[M,K](bf16) * B[N,K]^T(bf16) + bias, f32 accumulate.
// Block tile 128x256, 8 waves (wave32) in 2x4, each wave 64x64 via 4x4 WMMA
// 16x16 tiles (16 B/lane of LDS per WMMA).
// LDS: double buffered; A and B both stored K-major 32 bf16 (64B) per row.
// Stage = A:128 rows + B:256 rows -> (512 + 1024) v4u = 24 KB; x2 = 48 KB.

union Frag { v4u u[2]; v16bf v; };

__global__ __launch_bounds__(256)
void gemm_bf16_kernel(const unsigned short* __restrict__ xbf,   // [M][K]
                      const unsigned short* __restrict__ wbf,   // [N][K]
                      const float* __restrict__ bias,
                      float* __restrict__ out,
                      int M, int N, int K) {
  __shared__ v4u smem[3072];  // 2 stages * 1536 * 16B = 48 KB

  const int tid  = threadIdx.x;
  const int lane = tid & 31;
  const int w    = tid >> 5;
  const int wm   = (w >> 2) << 6;   // 0 or 64
  const int wn   = (w & 3) << 6;    // 0,64,128,192
  const int hf   = lane >> 4;       // half-wave
  const int ml   = lane & 15;

  const long bm = (long)blockIdx.y * 128;
  const long bn = (long)blockIdx.x * 256;
  const unsigned short* Ap = xbf + bm * K;
  const unsigned short* Bp = wbf + bn * K;

  v8f acc[16];
  {
    v8f z = {0.f, 0.f, 0.f, 0.f, 0.f, 0.f, 0.f, 0.f};
#pragma unroll
    for (int t = 0; t < 16; ++t) acc[t] = z;
  }

  auto stage = [&](int kt, int buf) {
    const int k0 = kt << 5;
    v4u* As = &smem[buf * 1536];
    v4u* Bs = As + 512;
#pragma unroll
    for (int i = 0; i < 2; ++i) {            // A: 128 rows x 4 chunks
      const int c = tid + (i << 8);          // 0..511
      const int row = c >> 2, seg = c & 3;
      const v4u* ga = (const v4u*)(Ap + (long)row * K + k0 + (seg << 3));
      v4u* la = As + row * 4 + seg;
#if USE_ASYNC_LDS
      __builtin_amdgcn_global_load_async_to_lds_b128(ASYNC_G(ga), ASYNC_L(la), 0, 0);
#else
      *la = *ga;
#endif
    }
#pragma unroll
    for (int i = 0; i < 4; ++i) {            // B: 256 rows x 4 chunks
      const int c = tid + (i << 8);          // 0..1023
      const int row = c >> 2, seg = c & 3;
      const v4u* gb = (const v4u*)(Bp + (long)row * K + k0 + (seg << 3));
      v4u* lb = Bs + row * 4 + seg;
#if USE_ASYNC_LDS
      __builtin_amdgcn_global_load_async_to_lds_b128(ASYNC_G(gb), ASYNC_L(lb), 0, 0);
#else
      *lb = *gb;
#endif
    }
  };

  stage(0, 0);
  WAIT_ASYNC();
  __syncthreads();

  const int NT = K >> 5;
  for (int kt = 0; kt < NT; ++kt) {
    const int cur = kt & 1;
    if (kt + 1 < NT) stage(kt + 1, cur ^ 1);

    const v4u* As = &smem[cur * 1536];
    const v4u* Bs = As + 512;

    Frag b[4];
#pragma unroll
    for (int tn = 0; tn < 4; ++tn) {
      const int n = wn + tn * 16 + ml;
      // B 32x16 bf16 frag: lane(n=ml, half=hf): K in [16hf, 16hf+16), contiguous
      b[tn].u[0] = Bs[n * 4 + 2 * hf];
      b[tn].u[1] = Bs[n * 4 + 2 * hf + 1];
    }
#pragma unroll
    for (int tm = 0; tm < 4; ++tm) {
      Frag a;
      const int r = wm + tm * 16 + ml;
      // A 16x32 bf16 frag: lane(m=ml, half=hf): K in [16hf, 16hf+16), contiguous
      a.u[0] = As[r * 4 + 2 * hf];
      a.u[1] = As[r * 4 + 2 * hf + 1];
#pragma unroll
      for (int tn = 0; tn < 4; ++tn)
        acc[tm * 4 + tn] = __builtin_amdgcn_wmma_f32_16x16x32_bf16(
            false, a.v, false, b[tn].v, (short)0, acc[tm * 4 + tn],
            false, false);
    }

    if (kt + 1 < NT) {
      WAIT_ASYNC();
      __syncthreads();
    }
  }

  // epilogue: C frag element r -> m = r + 8*hf, n = ml
#pragma unroll
  for (int tm = 0; tm < 4; ++tm)
#pragma unroll
    for (int tn = 0; tn < 4; ++tn) {
      const long col = bn + wn + tn * 16 + ml;
      const float bv = bias[col];
      float* op = out + (bm + wm + tm * 16 + hf * 8) * (long)N + col;
      const v8f c = acc[tm * 4 + tn];
#pragma unroll
      for (int r = 0; r < 8; ++r) op[(long)r * N] = c[r] + bv;
    }
}

// ---------------- launcher ----------------

extern "C" void kernel_launch(void* const* d_in, const int* in_sizes, int n_in,
                              void* d_out, int out_size, void* d_ws, size_t ws_size,
                              hipStream_t stream) {
  const float* x    = (const float*)d_in[0];
  const float* bw   = (const float*)d_in[1];
  const float* bias = (const float*)d_in[2];
  const float* spec = (const float*)d_in[3];
  const int*   idx  = (const int*)d_in[4];

  const int  D  = in_sizes[2];               // 4096
  const long M  = (long)in_sizes[0] / D;     // 4096
  const int  NF = in_sizes[3];               // 2000
  const int  K = D, N = D;

  char* ws = (char*)d_ws;
  float*          deltaT = (float*)ws;                                 // D*D*4 B
  unsigned short* wbf    = (unsigned short*)(ws + (size_t)D * D * 4);  // D*D*2 B
  unsigned short* xbf    = (unsigned short*)(ws + (size_t)D * D * 6);  // M*D*2 B

  const long n4 = (long)D * D / 4;
  zero_f4_kernel<<<(unsigned)((n4 + 255) / 256), 256, 0, stream>>>((v4f*)deltaT, n4);

  scatter_kernel<<<(NF + 255) / 256, 256, 0, stream>>>(idx, spec, deltaT, NF, D);

  const long n8w = (long)D * D / 8;
  build_weff_kernel<<<(unsigned)((n8w + 255) / 256), 256, 0, stream>>>(bw, deltaT, wbf, n8w);

  const long n8x = M * D / 8;
  cvt_x_kernel<<<(unsigned)((n8x + 255) / 256), 256, 0, stream>>>(x, xbf, n8x);

  dim3 grid(N / 256, (unsigned)(M / 128));
  gemm_bf16_kernel<<<grid, 256, 0, stream>>>(xbf, wbf, bias, (float*)d_out,
                                             (int)M, N, K);
}
